// HAN_16432544874760
// MI455X (gfx1250) — compile-verified, matrix operand order checked by reading, not run on hardware
//
#include <hip/hip_runtime.h>
#include <hip/hip_bf16.h>

// ---------------------------------------------------------------------------
// HAN (2-layer heterogeneous GAT) for MI455X / gfx1250, wave32 + WMMA.
// ---------------------------------------------------------------------------

typedef __attribute__((ext_vector_type(16))) _Float16 v16h;
typedef __attribute__((ext_vector_type(8)))  float    v8f;

#define TM 64     // block tile M
#define TN 128    // block tile N
#define TK 32     // K step (= WMMA K for f16)

// ---------------------------------------------------------------------------
// Tiled GEMM: Y[M,N] = X[M,K] @ W[K,N] + bias  (mode 1: tanh epilogue)
// f32 in memory, f16 in LDS, f32 accumulate via v_wmma_f32_16x16x32_f16.
// Block = 256 threads = 8 wave32 arranged 2x4; each wave computes a 32x32
// output region = 2x2 WMMA tiles -> 4 v_wmma per K-step with A/B fragment
// reuse. B tile is stored transposed in LDS so fragment K-pairs are
// contiguous and load as ds_load_b32.
// ---------------------------------------------------------------------------
__global__ __launch_bounds__(256)
void k_gemm_wmma(const float* __restrict__ X, const float* __restrict__ W,
                 const float* __restrict__ bias, float* __restrict__ Y,
                 int M, int K, int N, int mode)
{
    __shared__ _Float16 As[TM][TK + 8];      // [m][k]
    __shared__ _Float16 Bt[TN][TK + 8];      // [n][k]  (transposed tile)

    const int tid  = threadIdx.x;
    const int lane = tid & 31;
    const int wave = tid >> 5;     // 0..7
    const int wrow = wave >> 2;    // 0..1  -> M sub-region (32 rows)
    const int wcol = wave & 3;     // 0..3  -> N sub-region (32 cols)

    const int M0 = blockIdx.x * TM;
    const int N0 = blockIdx.y * TN;

    v8f acc00 = {}, acc01 = {}, acc10 = {}, acc11 = {};

    union H2  { unsigned u; _Float16 h[2]; };
    union Frag { v16h v; unsigned u[8]; };

    for (int k0 = 0; k0 < K; k0 += TK) {
        // Prefetch next K tile (global_prefetch_b8).
        if (k0 + TK < K) {
            int pr = M0 + (tid & 63);
            if (pr >= M) pr = M - 1;
            __builtin_prefetch(&X[(long)pr * K + k0 + TK], 0, 3);
            __builtin_prefetch(&W[(long)(k0 + TK + (tid >> 2)) * N + N0 + ((tid & 3) << 5)], 0, 3);
        }

        // Stage A tile (64x32) as float2 -> packed half2 (ds_store_b32).
        for (int t = tid; t < TM * (TK / 2); t += 256) {
            int r  = t >> 4;
            int c2 = (t & 15) * 2;
            int gr = M0 + r;
            float2 v = make_float2(0.f, 0.f);
            if (gr < M) v = *(const float2*)&X[(long)gr * K + k0 + c2];
            H2 pk; pk.h[0] = (_Float16)v.x; pk.h[1] = (_Float16)v.y;
            *(unsigned*)&As[r][c2] = pk.u;
        }
        // Stage B tile (32x128), transposing into Bt[n][k].
        for (int t = tid; t < TK * (TN / 2); t += 256) {
            int r  = t >> 6;            // k row 0..31
            int c2 = (t & 63) * 2;      // n pair
            float2 w2 = *(const float2*)&W[(long)(k0 + r) * N + N0 + c2];
            Bt[c2][r]     = (_Float16)w2.x;
            Bt[c2 + 1][r] = (_Float16)w2.y;
        }
        __syncthreads();

        // Build fragments (ISA 7.12.2 16-bit layout): lane = M%16 (or N%16),
        // K-pairs split across lane halves; each dword = halves (k, k+1).
        const int mr = lane & 15;
        const int kb = (lane >> 4) * 8;
        Frag a0, a1, b0, b1;
#pragma unroll
        for (int i = 0; i < 8; ++i) {
            int k = (i < 4) ? (kb + 2 * i) : (16 + kb + 2 * (i - 4));
            a0.u[i] = *(const unsigned*)&As[wrow * 32 + mr][k];
            a1.u[i] = *(const unsigned*)&As[wrow * 32 + 16 + mr][k];
            b0.u[i] = *(const unsigned*)&Bt[wcol * 32 + mr][k];
            b1.u[i] = *(const unsigned*)&Bt[wcol * 32 + 16 + mr][k];
        }

        acc00 = __builtin_amdgcn_wmma_f32_16x16x32_f16(false, a0.v, false, b0.v,
                                                       (short)0, acc00, false, false);
        acc01 = __builtin_amdgcn_wmma_f32_16x16x32_f16(false, a0.v, false, b1.v,
                                                       (short)0, acc01, false, false);
        acc10 = __builtin_amdgcn_wmma_f32_16x16x32_f16(false, a1.v, false, b0.v,
                                                       (short)0, acc10, false, false);
        acc11 = __builtin_amdgcn_wmma_f32_16x16x32_f16(false, a1.v, false, b1.v,
                                                       (short)0, acc11, false, false);
        __syncthreads();
    }

    // Epilogue. C/D layout: lane holds col = lane&15, rows i + 8*(lane>=16).
    const int lc = lane & 15;
    const int lr = (lane >> 4) * 8;
    const int rowb = M0 + wrow * 32;
    const int colb = N0 + wcol * 32;

    auto emit = [&](const v8f& a, int r0, int c0) {
        float bv = bias[c0];
#pragma unroll
        for (int i = 0; i < 8; ++i) {
            int row = r0 + i;
            if (row < M) {
                float v = a[i] + bv;
                if (mode == 1) v = tanhf(v);
                Y[(long)row * N + c0] = v;
            }
        }
    };
    emit(acc00, rowb + lr,      colb + lc);
    emit(acc01, rowb + lr,      colb + 16 + lc);
    emit(acc10, rowb + 16 + lr, colb + lc);
    emit(acc11, rowb + 16 + lr, colb + 16 + lc);
}

// ---------------------------------------------------------------------------
// Elementwise / edge-attention kernels
// ---------------------------------------------------------------------------
__global__ void k_fill(float* __restrict__ p, float v, long n) {
    long i = (long)blockIdx.x * 256 + threadIdx.x;
    if (i < n) p[i] = v;
}

__global__ void k_relu(float* __restrict__ p, long n) {
    long i = (long)blockIdx.x * 256 + threadIdx.x;
    if (i < n) p[i] = fmaxf(p[i], 0.f);
}

// al[n,h] = sum_d z[n, h*32+d] * att[h*32+d]
__global__ void k_scores(const float* __restrict__ z, const float* __restrict__ att,
                         float* __restrict__ al, int Nn)
{
    int i = blockIdx.x * 256 + threadIdx.x;
    if (i >= Nn * 8) return;
    int n = i >> 3, h = i & 7;
    const float* zp = z + (long)n * 256 + h * 32;
    const float* ap = att + h * 32;
    float s = 0.f;
#pragma unroll
    for (int d = 0; d < 32; ++d) s += zp[d] * ap[d];
    al[i] = s;
}

// alpha[e,h] = leaky_relu(als[s]+ald[d]);  m[d,h] = segment max (float atomicMax)
__global__ void k_alpha_max(const int* __restrict__ es, const int* __restrict__ ed,
                            const float* __restrict__ als, const float* __restrict__ ald,
                            float* __restrict__ alpha, float* __restrict__ m, int E)
{
    int i = blockIdx.x * 256 + threadIdx.x;
    if (i >= E * 8) return;
    int e = i >> 3, h = i & 7;
    int s = es[e], d = ed[e];
    float a = als[s * 8 + h] + ald[d * 8 + h];
    a = (a > 0.f) ? a : 0.2f * a;
    alpha[i] = a;
    atomicMax(&m[d * 8 + h], a);   // global_atomic_max_num_f32
}

// alpha[e,h] = exp(alpha - m[d]);  den[d,h] += alpha
__global__ void k_expsum(const int* __restrict__ ed, float* __restrict__ alpha,
                         const float* __restrict__ m, float* __restrict__ den, int E)
{
    int i = blockIdx.x * 256 + threadIdx.x;
    if (i >= E * 8) return;
    int e = i >> 3, h = i & 7;
    int d = ed[e];
    float v = __expf(alpha[i] - m[d * 8 + h]);
    alpha[i] = v;
    atomicAdd(&den[d * 8 + h], v);
}

// out[d, c] += (alpha[e,h]/den[d,h]) * z_src[s, c].  One block per edge:
// 256 lanes cover the full feature row -> coalesced gather + scatter.
__global__ void k_scatter(const int* __restrict__ es, const int* __restrict__ ed,
                          const float* __restrict__ alpha, const float* __restrict__ den,
                          const float* __restrict__ zsrc, float* __restrict__ out, int E)
{
    int e = blockIdx.x;
    int c = threadIdx.x;          // 0..255
    int h = c >> 5;
    int s = es[e], d = ed[e];
    float coef = alpha[e * 8 + h] / den[d * 8 + h];
    atomicAdd(&out[(long)d * 256 + c], coef * zsrc[(long)s * 256 + c]);
}

// w += mean_n( tmp[n,:] . q )  -- grid-stride + block tree + atomicAdd
__global__ void k_dotq_mean(const float* __restrict__ t, const float* __restrict__ q,
                            float* __restrict__ wout, int Nn)
{
    __shared__ float red[256];
    long total = (long)Nn * 256;
    float acc = 0.f;
    for (long i = (long)blockIdx.x * 256 + threadIdx.x; i < total;
         i += (long)gridDim.x * 256)
        acc += t[i] * q[i & 255];
    red[threadIdx.x] = acc;
    __syncthreads();
    for (int s = 128; s > 0; s >>= 1) {
        if (threadIdx.x < s) red[threadIdx.x] += red[threadIdx.x + s];
        __syncthreads();
    }
    if (threadIdx.x == 0) atomicAdd(wout, red[0] / (float)Nn);
}

// dst = relu( beta0*o0 + beta1*o1 ), beta = softmax(w[0..nouts-1])
__global__ void k_blend(const float* __restrict__ o0, const float* __restrict__ o1,
                        const float* __restrict__ w, float* __restrict__ dst,
                        long n, int nouts)
{
    long i = (long)blockIdx.x * 256 + threadIdx.x;
    if (i >= n) return;
    float b0 = 1.f, b1 = 0.f;
    if (nouts == 2) {
        float w0 = w[0], w1 = w[1];
        float mx = fmaxf(w0, w1);
        float e0 = __expf(w0 - mx), e1 = __expf(w1 - mx);
        float inv = 1.f / (e0 + e1);
        b0 = e0 * inv; b1 = e1 * inv;
    }
    float v = b0 * o0[i] + ((nouts == 2) ? b1 * o1[i] : 0.f);
    dst[i] = fmaxf(v, 0.f);
}

// logits[n,c] = a[n,:] . W[:,c] + b[c]   (N=2, scalar kernel is fine)
__global__ void k_linear(const float* __restrict__ a, const float* __restrict__ W,
                         const float* __restrict__ b, float* __restrict__ out, int Nn)
{
    int i = blockIdx.x * 256 + threadIdx.x;
    if (i >= Nn * 2) return;
    int n = i >> 1, c = i & 1;
    const float* ar = a + (long)n * 256;
    float s = b[c];
#pragma unroll 8
    for (int k = 0; k < 256; ++k) s += ar[k] * W[k * 2 + c];
    out[i] = s;
}

// ---------------------------------------------------------------------------
// Host orchestration
// ---------------------------------------------------------------------------
extern "C" void kernel_launch(void* const* d_in, const int* in_sizes, int n_in,
                              void* d_out, int out_size, void* d_ws, size_t ws_size,
                              hipStream_t stream)
{
    (void)in_sizes; (void)n_in; (void)out_size; (void)ws_size;

    const int NA = 50000, NT = 50000, E = 300000, FH = 256;

    const float* x_a  = (const float*)d_in[0];
    const float* x_t  = (const float*)d_in[1];
    const float* W1a  = (const float*)d_in[2];
    const float* b1a  = (const float*)d_in[3];
    const float* W1t  = (const float*)d_in[4];
    const float* b1t  = (const float*)d_in[5];
    const float* a1s  = (const float*)d_in[6];
    const float* a1d  = (const float*)d_in[7];
    const float* q1   = (const float*)d_in[8];
    const float* k1W  = (const float*)d_in[9];
    const float* k1b  = (const float*)d_in[10];
    const float* W2a  = (const float*)d_in[11];
    const float* b2a  = (const float*)d_in[12];
    const float* W2t  = (const float*)d_in[13];
    const float* b2t  = (const float*)d_in[14];
    const float* a2s  = (const float*)d_in[15];
    const float* a2d  = (const float*)d_in[16];
    const float* q2   = (const float*)d_in[17];
    const float* k2W  = (const float*)d_in[18];
    const float* k2b  = (const float*)d_in[19];
    const float* linW = (const float*)d_in[20];
    const float* linb = (const float*)d_in[21];
    const int* ei_at  = (const int*)d_in[22];
    const int* ei_ta  = (const int*)d_in[23];
    const int* ei_aa  = (const int*)d_in[24];

    // Workspace carve-up
    float* p   = (float*)d_ws;
    float* z_a = p;  p += (size_t)NA * FH;
    float* z_t = p;  p += (size_t)NT * FH;
    float* z2a = p;  p += (size_t)NA * FH;
    float* z2t = p;  p += (size_t)NT * FH;
    float* out0= p;  p += (size_t)NT * FH;   // addr->tx output
    float* out1= p;  p += (size_t)NA * FH;   // tx->addr output
    float* out2= p;  p += (size_t)NA * FH;   // addr->addr output
    float* tmp = p;  p += (size_t)NA * FH;   // semantic tanh GEMM scratch
    float* als = p;  p += (size_t)NA * 8;
    float* ald = p;  p += (size_t)NA * 8;
    float* mb  = p;  p += (size_t)NA * 8;
    float* den = p;  p += (size_t)NA * 8;
    float* alp = p;  p += (size_t)E * 8;
    float* wbf = p;  p += 8;

    auto gemm = [&](const float* X, const float* W, const float* B, float* Y,
                    int M, int K, int N, int mode) {
        dim3 g((M + TM - 1) / TM, N / TN);
        k_gemm_wmma<<<g, 256, 0, stream>>>(X, W, B, Y, M, K, N, mode);
    };
    auto fill = [&](float* ptr, float v, long n) {
        k_fill<<<(int)((n + 255) / 256), 256, 0, stream>>>(ptr, v, n);
    };
    auto edge_attn = [&](const float* zs, const float* zd, const int* ei,
                         const float* as_, const float* ad_, int Ns, int Nd,
                         float* out) {
        fill(out, 0.f, (long)Nd * FH);
        fill(mb, -1e30f, (long)Nd * 8);
        fill(den, 0.f, (long)Nd * 8);
        k_scores<<<(Ns * 8 + 255) / 256, 256, 0, stream>>>(zs, as_, als, Ns);
        k_scores<<<(Nd * 8 + 255) / 256, 256, 0, stream>>>(zd, ad_, ald, Nd);
        k_alpha_max<<<(E * 8 + 255) / 256, 256, 0, stream>>>(ei, ei + E, als, ald, alp, mb, E);
        k_expsum<<<(E * 8 + 255) / 256, 256, 0, stream>>>(ei + E, alp, mb, den, E);
        k_scatter<<<E, 256, 0, stream>>>(ei, ei + E, alp, den, zs, out, E);
        k_relu<<<(int)(((long)Nd * FH + 255) / 256), 256, 0, stream>>>(out, (long)Nd * FH);
    };
    auto semantic = [&](const float* o0, const float* o1, int nouts,
                        const float* kW, const float* kb, const float* q,
                        int Nn, float* dst) {
        fill(wbf, 0.f, 8);
        gemm(o0, kW, kb, tmp, Nn, FH, FH, 1);
        k_dotq_mean<<<512, 256, 0, stream>>>(tmp, q, wbf + 0, Nn);
        if (nouts == 2) {
            gemm(o1, kW, kb, tmp, Nn, FH, FH, 1);
            k_dotq_mean<<<512, 256, 0, stream>>>(tmp, q, wbf + 1, Nn);
        }
        long n = (long)Nn * FH;
        k_blend<<<(int)((n + 255) / 256), 256, 0, stream>>>(
            o0, (nouts == 2) ? o1 : o0, wbf, dst, n, nouts);
    };

    // ---------------- Layer 1 ----------------
    gemm(x_a, W1a, b1a, z_a, NA, 128, FH, 0);
    gemm(x_t, W1t, b1t, z_t, NT, 128, FH, 0);
    edge_attn(z_a, z_t, ei_at, a1s + 0,   a1d + 0,   NA, NT, out0);
    edge_attn(z_t, z_a, ei_ta, a1s + 256, a1d + 256, NT, NA, out1);
    edge_attn(z_a, z_a, ei_aa, a1s + 512, a1d + 512, NA, NA, out2);
    semantic(out0, nullptr, 1, k1W, k1b, q1, NT, z_t);   // new_t (relu fused)
    semantic(out1, out2,    2, k1W, k1b, q1, NA, z_a);   // new_a (relu fused)

    // ---------------- Layer 2 ----------------
    gemm(z_a, W2a, b2a, z2a, NA, FH, FH, 0);
    gemm(z_t, W2t, b2t, z2t, NT, FH, FH, 0);
    edge_attn(z2a, z2t, ei_at, a2s + 0,   a2d + 0,   NA, NT, out0); // new_t unused by output
    edge_attn(z2t, z2a, ei_ta, a2s + 256, a2d + 256, NT, NA, out1);
    edge_attn(z2a, z2a, ei_aa, a2s + 512, a2d + 512, NA, NA, out2);
    semantic(out1, out2, 2, k2W, k2b, q2, NA, z_a);      // new_a + relu

    // ---------------- Classifier ----------------
    k_linear<<<(NA * 2 + 255) / 256, 256, 0, stream>>>(z_a, linW, linb,
                                                       (float*)d_out, NA);
}
